// Multi_Head_Attention_60610578482075
// MI455X (gfx1250) — matrix operand order repaired and checked by reading, not compiled, not run on hardware
//
#include <hip/hip_runtime.h>

// ---------------------------------------------------------------------------
// Multi-head attention, bf16 WMMA + TDM pipeline for gfx1250 (MI455X).
//   B=2, S=4096, D_MODEL=512, H=8, d_k=d_v=64
// Stage 1: Q/K/V projections (3 GEMMs 8192x512x512, f32->bf16).
//          V is stored TRANSPOSED [b,h,d,s] so attention V tiles are
//          straight 2-D TDM copies into the [d][key] LDS layout.
// Stage 2: flash attention per (b,h), 128-query tiles, online softmax.
//          K/V tiles staged by the Tensor Data Mover (tensor_load_to_lds).
// Stage 3: output projection (GEMM 8192x512x512, bf16 A, f32 out)
// ---------------------------------------------------------------------------

#define S_LEN   4096
#define DMODEL  512
#define NHEAD   8
#define DHEAD   64
#define NBATCH  2
#define MROWS   (NBATCH * S_LEN)   // 8192

typedef __attribute__((ext_vector_type(16))) __bf16 v16bf;
typedef __attribute__((ext_vector_type(8)))  float  v8f;
typedef __attribute__((ext_vector_type(4)))  unsigned int v4u;
typedef __attribute__((ext_vector_type(8)))  int    v8i;
typedef __attribute__((ext_vector_type(4)))  int    v4i;

union FragAB { v16bf v; unsigned int d[8]; unsigned short u[16]; };

__device__ __forceinline__ unsigned short f32_to_bf16(float f) {
    union { float f; unsigned int u; } x; x.f = f;
    unsigned int r = x.u + 0x7FFFu + ((x.u >> 16) & 1u);  // round-nearest-even
    return (unsigned short)(r >> 16);
}

__device__ __forceinline__ v8f v8f_zero() {
    v8f z = {0.f, 0.f, 0.f, 0.f, 0.f, 0.f, 0.f, 0.f};
    return z;
}

// A-matrix fragment (16x32 bf16, MxK). Storage `p` is row-major [row][k],
// stride in ushorts (must be even). Lanes 0-15: row=lane, K={0..7,16..23};
// lanes 16-31: row=lane-16, K={8..15,24..31}.  (ISA 7.12.2)
__device__ __forceinline__ v16bf frag_load_a(const unsigned short* p, int stride,
                                             int row0, int k0, int lane) {
    FragAB f;
    const unsigned short* r = p + (row0 + (lane & 15)) * stride + k0 + ((lane >> 4) << 3);
#pragma unroll
    for (int j = 0; j < 4; ++j)
        f.d[j] = *(const unsigned int*)(r + 2 * j);
#pragma unroll
    for (int j = 0; j < 4; ++j)
        f.d[4 + j] = *(const unsigned int*)(r + 16 + 2 * j);
    return f.v;
}

// B-matrix fragment (32x16 bf16, KxN). Storage `p` is [n][k]: each stored row
// holds all K for one output column n.  Lane holds column n = n0 + lane%16;
// lanes 0-15: K={0..7,16..23}, lanes 16-31: K={8..15,24..31}.
__device__ __forceinline__ v16bf frag_load_b(const unsigned short* p, int stride,
                                             int n0, int k0, int lane) {
    FragAB f;
    const unsigned short* r = p + (n0 + (lane & 15)) * stride + k0 + ((lane >> 4) << 3);
#pragma unroll
    for (int j = 0; j < 4; ++j)
        f.d[j] = *(const unsigned int*)(r + 2 * j);
#pragma unroll
    for (int j = 0; j < 4; ++j)
        f.d[4 + j] = *(const unsigned int*)(r + 16 + 2 * j);
    return f.v;
}

__device__ __forceinline__ v8f wmma_bf16(v16bf a, v16bf b, v8f c) {
    // D = A(16x32) * B(32x16) + C, f32 accumulate
    return __builtin_amdgcn_wmma_f32_16x16x32_bf16(
        false, a, false, b, (short)0, c, false, false);
}

// ---------------------------------------------------------------------------
// Tensor Data Mover: 2-D tile load, bf16 (data_size=2B) elements.
//   tile_w elements contiguous along x, tile_h rows, row stride stride_e
//   elements.  LDS receives the tile packed row-major [tile_h][tile_w].
// Descriptor bitfields per CDNA5 ISA.
// ---------------------------------------------------------------------------
__device__ __forceinline__ void tdm_load_2d_bf16(unsigned int lds_addr,
                                                 const void* gptr,
                                                 int tile_w, int tile_h,
                                                 long long stride_e) {
    unsigned long long ga = (unsigned long long)gptr;
    v4u g0;
    g0[0] = 1u;                                    // count=1, user mode
    g0[1] = lds_addr;                              // lds_addr[31:0]
    g0[2] = (unsigned int)(ga & 0xFFFFFFFFu);      // global_addr[31:0]
    g0[3] = (unsigned int)((ga >> 32) & 0x01FFFFFFu) | 0x80000000u; // type=2

    unsigned int tw = (unsigned int)tile_w, th = (unsigned int)tile_h;
    unsigned long long st = (unsigned long long)stride_e;
    v8i g1;
    g1[0] = (int)(1u << 16);                       // data_size=1 (2 bytes)
    g1[1] = (int)((tw & 0xFFFFu) << 16);           // tensor_dim0[15:0]
    g1[2] = (int)(((tw >> 16) & 0xFFFFu) | ((th & 0xFFFFu) << 16)); // dim0 hi | dim1 lo
    g1[3] = (int)(((th >> 16) & 0xFFFFu) | ((tw & 0xFFFFu) << 16)); // dim1 hi | tile_dim0
    g1[4] = (int)(th & 0xFFFFu);                   // tile_dim1 (tile_dim2=0)
    g1[5] = (int)(unsigned int)(st & 0xFFFFFFFFull);        // dim0_stride[31:0]
    g1[6] = (int)(unsigned int)((st >> 32) & 0xFFFFull);    // dim0_stride[47:32]
    g1[7] = 0;

    v4i z4 = {0, 0, 0, 0};
#if __has_include(<hip/amd_detail/amd_gfx1250_TDM.h>)
    v8i z8 = {0, 0, 0, 0, 0, 0, 0, 0};
    __builtin_amdgcn_tensor_load_to_lds(g0, g1, z4, z4, z8, 0);   // clang-23, 6-arg
#else
    __builtin_amdgcn_tensor_load_to_lds(g0, g1, z4, z4, 0);       // ROCm 7.2, 5-arg
#endif
}

// ---------------------------------------------------------------------------
// Stage 1: projection GEMM.  C[m,n] = sum_k X[m,k] * W[n,k]
// X: [MROWS, 512] f32.  W: [512, 512] f32 ([h*64+k_out, d_in]).
// v_transpose==0: store bf16 into [b, h, s, 64]   (Q, K)
// v_transpose==1: store bf16 into [b, h, 64, s]   (V, TDM/B-frag friendly)
// ---------------------------------------------------------------------------
__global__ void __launch_bounds__(256)
proj_gemm_kernel(const float* __restrict__ X, const float* __restrict__ W,
                 unsigned short* __restrict__ dst, int v_transpose) {
    __shared__ unsigned short As[128 * 64];
    __shared__ unsigned short Bs[128 * 64];

    const int tid  = threadIdx.x;
    const int lane = tid & 31;
    const int wave = tid >> 5;
    const int wr = wave >> 2;   // 0..1
    const int wc = wave & 3;    // 0..3
    const int m0 = blockIdx.x * 128;
    const int n0 = blockIdx.y * 128;

    v8f acc[4][2];
#pragma unroll
    for (int mt = 0; mt < 4; ++mt)
#pragma unroll
        for (int nt = 0; nt < 2; ++nt) acc[mt][nt] = v8f_zero();

    for (int k0 = 0; k0 < DMODEL; k0 += 64) {
        __syncthreads();
        // stage A (128x64) and B (128x64), f32 -> bf16
        for (int idx = tid; idx < 128 * 64; idx += 256) {
            int r = idx >> 6, c = idx & 63;
            As[idx] = f32_to_bf16(X[(m0 + r) * DMODEL + k0 + c]);
            Bs[idx] = f32_to_bf16(W[(n0 + r) * DMODEL + k0 + c]);
        }
        __syncthreads();
#pragma unroll
        for (int kk = 0; kk < 64; kk += 32) {
            v16bf af[4];
#pragma unroll
            for (int mt = 0; mt < 4; ++mt)
                af[mt] = frag_load_a(As, 64, wr * 64 + mt * 16, kk, lane);
            v16bf bfm[2];
#pragma unroll
            for (int nt = 0; nt < 2; ++nt)
                bfm[nt] = frag_load_b(Bs, 64, wc * 32 + nt * 16, kk, lane);
#pragma unroll
            for (int mt = 0; mt < 4; ++mt)
#pragma unroll
                for (int nt = 0; nt < 2; ++nt)
                    acc[mt][nt] = wmma_bf16(af[mt], bfm[nt], acc[mt][nt]);
        }
    }

#pragma unroll
    for (int mt = 0; mt < 4; ++mt)
#pragma unroll
        for (int nt = 0; nt < 2; ++nt)
#pragma unroll
            for (int i = 0; i < 8; ++i) {
                int m = m0 + wr * 64 + mt * 16 + i + ((lane >> 4) << 3);
                int n = n0 + wc * 32 + nt * 16 + (lane & 15);
                int b = m >> 12, s = m & (S_LEN - 1);
                int h = n >> 6, k = n & (DHEAD - 1);
                unsigned short val = f32_to_bf16(acc[mt][nt][i]);
                if (v_transpose)
                    dst[(((b << 3) + h) * DHEAD + k) * S_LEN + s] = val;
                else
                    dst[(((b << 3) + h) * S_LEN + s) * DHEAD + k] = val;
            }
}

// ---------------------------------------------------------------------------
// Stage 2: flash attention for one (b, h, 128-query tile).
// 8 waves, each owns 16 query rows. K tile in LDS as [key][d] (B-frag ready
// for Q*K^T), V tile in LDS as [d][key] (B-frag ready for P*V); both staged
// by the Tensor Data Mover.
// ---------------------------------------------------------------------------
__global__ void __launch_bounds__(256)
attn_kernel(const unsigned short* __restrict__ Qh,
            const unsigned short* __restrict__ Kh,
            const unsigned short* __restrict__ Vt,
            unsigned short* __restrict__ Cc) {
    __shared__ unsigned short Ks[128 * 64];       // [key][d]
    __shared__ unsigned short Vs[64 * 128];       // [d][key]
    __shared__ unsigned short Ps[8][16 * 128];    // per-wave P tile [q][key]

    const int tid  = threadIdx.x;
    const int lane = tid & 31;
    const int wave = tid >> 5;
    const int qt = blockIdx.x;     // 0..31
    const int h  = blockIdx.y;     // 0..7
    const int b  = blockIdx.z;     // 0..1

    const unsigned short* Qbase = Qh + ((((b << 3) + h) * S_LEN) + qt * 128) * DHEAD;
    const unsigned short* Kbase = Kh + (((b << 3) + h) * S_LEN) * DHEAD;     // [s][d]
    const unsigned short* Vbase = Vt + (((b << 3) + h) * DHEAD) * S_LEN;     // [d][s]

    const unsigned int ldsK = (unsigned int)(unsigned long long)(const void*)Ks;
    const unsigned int ldsV = (unsigned int)(unsigned long long)(const void*)Vs;

    // Q fragments for this wave's 16 rows, loaded straight from global
    v16bf qf[2];
#pragma unroll
    for (int kk = 0; kk < 2; ++kk)
        qf[kk] = frag_load_a(Qbase, DHEAD, wave * 16, kk * 32, lane);

    float mrow[8], lrow[8];
#pragma unroll
    for (int i = 0; i < 8; ++i) { mrow[i] = -3.0e38f; lrow[i] = 0.f; }
    v8f o[4];
#pragma unroll
    for (int nt = 0; nt < 4; ++nt) o[nt] = v8f_zero();

    for (int s0 = 0; s0 < S_LEN; s0 += 128) {
        __syncthreads();  // previous tile's readers done before TDM overwrites
        if (wave == 0) {
            // K tile: 128 keys x 64 d, contiguous rows of 64 elems
            tdm_load_2d_bf16(ldsK, Kbase + s0 * DHEAD, 64, 128, 64);
            // V tile: 64 d-rows x 128 keys, row stride S_LEN
            tdm_load_2d_bf16(ldsV, Vbase + s0, 128, 64, S_LEN);
            if (s0 + 128 < S_LEN) {   // prefetch next tiles toward L2
                __builtin_prefetch(Kbase + (s0 + 128) * DHEAD, 0, 1);
                __builtin_prefetch(Vbase + (s0 + 128), 0, 1);
            }
            __builtin_amdgcn_s_wait_tensorcnt(0);
        }
        __syncthreads();  // LDS tiles ready for all waves

        // scores: 16 q-rows x 128 keys  (8 fragments, K=64 chained)
        v8f sc[8];
#pragma unroll
        for (int f = 0; f < 8; ++f) {
            v8f c = v8f_zero();
            v16bf b0 = frag_load_b(Ks, 64, f * 16, 0, lane);
            c = wmma_bf16(qf[0], b0, c);
            v16bf b1 = frag_load_b(Ks, 64, f * 16, 32, lane);
            c = wmma_bf16(qf[1], b1, c);
            sc[f] = c * 0.125f;   // 1/sqrt(d_k)
        }

        // online softmax update (rows live in half-wave lane groups of 16)
#pragma unroll
        for (int i = 0; i < 8; ++i) {
            float t = sc[0][i];
#pragma unroll
            for (int f = 1; f < 8; ++f) t = fmaxf(t, sc[f][i]);
#pragma unroll
            for (int off = 1; off < 16; off <<= 1)
                t = fmaxf(t, __shfl_xor(t, off, 16));
            float mnew = fmaxf(mrow[i], t);
            float alpha = __expf(mrow[i] - mnew);
            mrow[i] = mnew;
            lrow[i] *= alpha;
#pragma unroll
            for (int nt = 0; nt < 4; ++nt) o[nt][i] *= alpha;
        }

        // P = exp(sc - m), row sums, store P (bf16) to per-wave LDS slab
        float psum[8];
#pragma unroll
        for (int i = 0; i < 8; ++i) psum[i] = 0.f;
        unsigned short* pw = &Ps[wave][0];
#pragma unroll
        for (int f = 0; f < 8; ++f)
#pragma unroll
            for (int i = 0; i < 8; ++i) {
                float p = __expf(sc[f][i] - mrow[i]);
                psum[i] += p;
                pw[(i + ((lane >> 4) << 3)) * 128 + f * 16 + (lane & 15)] =
                    f32_to_bf16(p);
            }
#pragma unroll
        for (int i = 0; i < 8; ++i) {
#pragma unroll
            for (int off = 1; off < 16; off <<= 1)
                psum[i] += __shfl_xor(psum[i], off, 16);
            lrow[i] += psum[i];
        }

        // intra-wave LDS store->load hazard on Ps
        asm volatile("s_wait_dscnt 0x0" ::: "memory");

        // O += P(16x128) * V(128x64): 4 output tiles, K=128 chained
        v16bf pf[4];
#pragma unroll
        for (int kk = 0; kk < 4; ++kk)
            pf[kk] = frag_load_a(pw, 128, 0, kk * 32, lane);
#pragma unroll
        for (int nt = 0; nt < 4; ++nt) {
            v8f c = o[nt];
#pragma unroll
            for (int kk = 0; kk < 4; ++kk) {
                v16bf vb = frag_load_b(Vs, 128, nt * 16, kk * 32, lane);
                c = wmma_bf16(pf[kk], vb, c);
            }
            o[nt] = c;
        }
    }

    // epilogue: O /= l, write bf16 concat [b, s, h*64 + v]
#pragma unroll
    for (int nt = 0; nt < 4; ++nt)
#pragma unroll
        for (int i = 0; i < 8; ++i) {
            float val = o[nt][i] / lrow[i];
            int q = qt * 128 + wave * 16 + i + ((lane >> 4) << 3);
            int v = nt * 16 + (lane & 15);
            Cc[(b * S_LEN + q) * DMODEL + h * DHEAD + v] = f32_to_bf16(val);
        }
}

// ---------------------------------------------------------------------------
// Stage 3: output GEMM.  out[m,n] = sum_f concat[m,f] * Wo[n,f]
// A: bf16 concat [MROWS, 512] (ws). B: f32 Wo [512, 512]. C: f32 d_out.
// ---------------------------------------------------------------------------
__global__ void __launch_bounds__(256)
out_gemm_kernel(const unsigned short* __restrict__ A,
                const float* __restrict__ Wo,
                float* __restrict__ out) {
    __shared__ unsigned short As[128 * 64];
    __shared__ unsigned short Bs[128 * 64];

    const int tid  = threadIdx.x;
    const int lane = tid & 31;
    const int wave = tid >> 5;
    const int wr = wave >> 2;
    const int wc = wave & 3;
    const int m0 = blockIdx.x * 128;
    const int n0 = blockIdx.y * 128;

    v8f acc[4][2];
#pragma unroll
    for (int mt = 0; mt < 4; ++mt)
#pragma unroll
        for (int nt = 0; nt < 2; ++nt) acc[mt][nt] = v8f_zero();

    for (int k0 = 0; k0 < DMODEL; k0 += 64) {
        __syncthreads();
        {
            uint4* ad = (uint4*)As;
            for (int idx = tid; idx < 1024; idx += 256) {
                int r = idx >> 3, c8 = idx & 7;
                ad[idx] = ((const uint4*)A)[(((m0 + r) * DMODEL + k0) >> 3) + c8];
            }
        }
        for (int idx = tid; idx < 128 * 64; idx += 256) {
            int r = idx >> 6, c = idx & 63;
            Bs[idx] = f32_to_bf16(Wo[(n0 + r) * DMODEL + k0 + c]);
        }
        __syncthreads();
#pragma unroll
        for (int kk = 0; kk < 64; kk += 32) {
            v16bf af[4];
#pragma unroll
            for (int mt = 0; mt < 4; ++mt)
                af[mt] = frag_load_a(As, 64, wr * 64 + mt * 16, kk, lane);
            v16bf bfm[2];
#pragma unroll
            for (int nt = 0; nt < 2; ++nt)
                bfm[nt] = frag_load_b(Bs, 64, wc * 32 + nt * 16, kk, lane);
#pragma unroll
            for (int mt = 0; mt < 4; ++mt)
#pragma unroll
                for (int nt = 0; nt < 2; ++nt)
                    acc[mt][nt] = wmma_bf16(af[mt], bfm[nt], acc[mt][nt]);
        }
    }

#pragma unroll
    for (int mt = 0; mt < 4; ++mt)
#pragma unroll
        for (int nt = 0; nt < 2; ++nt)
#pragma unroll
            for (int i = 0; i < 8; ++i) {
                int m = m0 + wr * 64 + mt * 16 + i + ((lane >> 4) << 3);
                int n = n0 + wc * 32 + nt * 16 + (lane & 15);
                out[m * DMODEL + n] = acc[mt][nt][i];
            }
}

// ---------------------------------------------------------------------------
extern "C" void kernel_launch(void* const* d_in, const int* in_sizes, int n_in,
                              void* d_out, int out_size, void* d_ws, size_t ws_size,
                              hipStream_t stream) {
    const float* Qin = (const float*)d_in[0];
    const float* Kin = (const float*)d_in[1];
    const float* Vin = (const float*)d_in[2];
    const float* Wq  = (const float*)d_in[3];  // [H,64,512] == [512,512] row-major
    const float* Wk  = (const float*)d_in[4];
    const float* Wv  = (const float*)d_in[5];
    const float* Wo  = (const float*)d_in[6];  // [512,512]

    // workspace carve (bf16 as ushort): Qh | Kh | Vt | concat  = 32 MB total
    const size_t elems = (size_t)NBATCH * NHEAD * S_LEN * DHEAD;  // 4,194,304
    unsigned short* qh = (unsigned short*)d_ws;
    unsigned short* kh = qh + elems;
    unsigned short* vt = kh + elems;
    unsigned short* cc = vt + elems;

    dim3 block(256);
    dim3 gproj(MROWS / 128, DMODEL / 128);          // (64, 4)
    proj_gemm_kernel<<<gproj, block, 0, stream>>>(Qin, Wq, qh, 0);
    proj_gemm_kernel<<<gproj, block, 0, stream>>>(Kin, Wk, kh, 0);
    proj_gemm_kernel<<<gproj, block, 0, stream>>>(Vin, Wv, vt, 1);  // [b,h,d,s]

    dim3 gattn(S_LEN / 128, NHEAD, NBATCH);         // (32, 8, 2)
    attn_kernel<<<gattn, block, 0, stream>>>(qh, kh, vt, cc);

    out_gemm_kernel<<<gproj, block, 0, stream>>>(cc, Wo, (float*)d_out);
}
